// RNN_25701084299982
// MI455X (gfx1250) — matrix operand order repaired and checked by reading, not compiled, not run on hardware
//
#include <hip/hip_runtime.h>

typedef __attribute__((ext_vector_type(16))) _Float16 v16h;
typedef __attribute__((ext_vector_type(8)))  float    v8f;
typedef __attribute__((ext_vector_type(8)))  unsigned v8u;

constexpr int B  = 16;
constexpr int T  = 2048;
constexpr int H  = 1024;
constexpr int NI = 16;
constexpr int NO = 16;

constexpr int NB   = 16;      // workgroups (== NO, each WG also owns one z column)
constexpr int COLS = H / NB;  // 64 output neurons per WG
constexpr int NT   = 128;     // 4 waves per WG; wave w owns N-tile w (16 cols)
constexpr int KCH  = H / 32;  // 32 K-chunks of 32 for 16x16x32 WMMA

constexpr float DT      = 0.001f;
constexpr float DT_TAU  = DT / 0.01f;       // 0.1
constexpr float I_TAU_D = 1.0f / 0.2f;      // 5.0
constexpr float I_TAU_F = 1.0f / 1.5f;
constexpr float BETA    = 18.0f;
constexpr float GAIN    = 2.2f / 0.5f;
constexpr float AG      = 8.0f;
constexpr float ATH     = 0.5f;
constexpr float NDEC    = 1.0f - DT / 0.01f;  // noise decay (noise_std=0 -> stays 0)

__device__ __forceinline__ float sigm(float v) { return 1.0f / (1.0f + __expf(-v)); }

__global__ void init_bar(unsigned* bar) {
  if (threadIdx.x < 2) bar[threadIdx.x] = 0u;
}

__device__ __forceinline__ void grid_barrier(unsigned* bar) {
  __syncthreads();
  __builtin_amdgcn_fence(__ATOMIC_RELEASE, "agent");
  if (threadIdx.x == 0) {
    unsigned g0 = __hip_atomic_load(&bar[1], __ATOMIC_RELAXED, __HIP_MEMORY_SCOPE_AGENT);
    unsigned old = __hip_atomic_fetch_add(&bar[0], 1u, __ATOMIC_ACQ_REL, __HIP_MEMORY_SCOPE_AGENT);
    if (old == (unsigned)NB - 1u) {
      __hip_atomic_store(&bar[0], 0u, __ATOMIC_RELAXED, __HIP_MEMORY_SCOPE_AGENT);
      __hip_atomic_fetch_add(&bar[1], 1u, __ATOMIC_RELEASE, __HIP_MEMORY_SCOPE_AGENT);
    } else {
      while (__hip_atomic_load(&bar[1], __ATOMIC_ACQUIRE, __HIP_MEMORY_SCOPE_AGENT) == g0)
        __builtin_amdgcn_s_sleep(8);
    }
  }
  __syncthreads();
  __builtin_amdgcn_fence(__ATOMIC_ACQUIRE, "agent");
}

__global__ __launch_bounds__(NT, 1)
void rnn_stp_persistent(const float* __restrict__ x,    const float* __restrict__ h0,
                        const float* __restrict__ r0,   const float* __restrict__ u0,
                        const float* __restrict__ Wih,  const float* __restrict__ Whh,
                        const float* __restrict__ Wmk,  const float* __restrict__ prl,
                        const float* __restrict__ psc,  const float* __restrict__ Whz,
                        const float* __restrict__ offs,
                        float* __restrict__ o_ext, float* __restrict__ o_h,
                        float* __restrict__ o_r,   float* __restrict__ o_u,
                        float* __restrict__ o_z,
                        _Float16* __restrict__ sbuf,   // 2 x [B][H] f16 ping-pong
                        float*    __restrict__ htbuf,  // 2 x [B][H] f32 ping-pong
                        unsigned* __restrict__ bar) {
  // LDS: weights pre-swizzled in WMMA B-fragment layout (f16 pairs), 128KB,
  // plus small staging buffers. Total ~138KB of the WGP's 320KB.
  __shared__ unsigned w_lds[4 * KCH * 32 * 8];  // [ntile][kc][lane][j] packed 2xf16
  __shared__ float    rec_lds[B * COLS];
  __shared__ float    x_lds[B * NI];
  __shared__ float    wih_lds[COLS * NI];
  __shared__ float    zred[NT];

  const int g    = blockIdx.x;
  const int tid  = threadIdx.x;
  const int wave = tid >> 5;
  const int lane = tid & 31;

  // ---- Preamble: swizzle W_hh slice (rows g*64..g*64+63) into B fragments ----
  for (int idx = tid; idx < 4 * KCH * 32 * 8; idx += NT) {
    const int j     = idx & 7;
    const int lane_ = (idx >> 3) & 31;
    const int kc    = (idx >> 8) & (KCH - 1);
    const int nt    = idx >> 13;
    const int n     = lane_ & 15;
    const int k0    = kc * 32 + ((lane_ >= 16) ? 16 : 0) + 2 * j;
    const int row   = g * COLS + nt * 16 + n;
    const size_t ix = (size_t)row * H + k0;
    const float v0  = GAIN * Whh[ix] * Wmk[ix];
    const float v1  = GAIN * Whh[ix + 1] * Wmk[ix + 1];
    const unsigned p0 = (unsigned)__builtin_bit_cast(unsigned short, (_Float16)v0);
    const unsigned p1 = (unsigned)__builtin_bit_cast(unsigned short, (_Float16)v1);
    w_lds[idx] = p0 | (p1 << 16);
  }
  for (int idx = tid; idx < COLS * NI; idx += NT) {
    const int c0 = idx >> 4, i = idx & 15;
    wih_lds[idx] = Wih[(size_t)(g * COLS + c0) * NI + i];
  }

  // ---- Per-thread state: 8 neurons-per-batch elements each ----
  float rst[8], ust[8], hst[8], htst[8], nst[8], prl_[8], psc_[8], bprl[8];
#pragma unroll
  for (int k2 = 0; k2 < 8; k2++) {
    const int e = tid + k2 * NT;
    const int b = e >> 6, c0 = e & 63;
    const int colg  = g * COLS + c0;
    const size_t ix = (size_t)b * H + colg;
    rst[k2]  = r0[ix];
    ust[k2]  = u0[ix];
    hst[k2]  = h0[ix];
    nst[k2]  = 0.0f;
    prl_[k2] = prl[colg];
    psc_[k2] = psc[colg];
    bprl[k2] = BETA * prl_[k2];
    htst[k2] = sigm(AG * (hst[k2] - ATH));
    sbuf[ix] = (_Float16)(rst[k2] * ust[k2] * psc_[k2] * htst[k2]);  // s_0, parity 0
  }

  grid_barrier(bar);

  const int m    = lane & 15;               // A-matrix row (batch index)
  const int ahl  = (lane >= 16) ? 4 : 0;    // half the A dword offset for K-half
  const int mb   = (lane >> 4) * 8;         // C-matrix M base
  const int nloc = wave * 16 + (lane & 15); // local output column of this lane

  for (int t = 0; t < T; t++) {
    const int p = t & 1;
    const unsigned* s32 = (const unsigned*)sbuf + (size_t)p * (B * H / 2);

    // stage x_t
    for (int idx = tid; idx < B * NI; idx += NT) {
      const int b = idx >> 4, i = idx & 15;
      x_lds[idx] = x[((size_t)b * T + t) * NI + i];
    }

    // ---- recurrent matmul: rec[:, wave-tile] = s_t @ Wm.T via WMMA ----
    {
      v8f c = {};
      const unsigned* srow = s32 + (size_t)m * (H / 2);
      for (int kc = 0; kc < KCH; kc++) {
        v8u au, bu;
        const unsigned* wp = &w_lds[(((wave * KCH) + kc) * 32 + lane) * 8];
#pragma unroll
        for (int j = 0; j < 8; j++)
          au[j] = srow[kc * 16 + 2 * ahl + j];
#pragma unroll
        for (int j = 0; j < 8; j++)
          bu[j] = wp[j];
        c = __builtin_amdgcn_wmma_f32_16x16x32_f16(
                false, __builtin_bit_cast(v16h, au),
                false, __builtin_bit_cast(v16h, bu),
                (short)0, c, false, false);
      }
#pragma unroll
      for (int i = 0; i < 8; i++)
        rec_lds[(mb + i) * COLS + nloc] = c[i];
    }
    __syncthreads();

    // ---- elementwise STP/Euler updates + outputs (no divisions) ----
    _Float16* sbW = sbuf  + (size_t)(p ^ 1) * (B * H);
    float*    htW = htbuf + (size_t)(p ^ 1) * (B * H);
#pragma unroll
    for (int k2 = 0; k2 < 8; k2++) {
      const int e = tid + k2 * NT;
      const int b = e >> 6, c0 = e & 63;
      const int colg = g * COLS + c0;
      float ext = nst[k2];
#pragma unroll
      for (int i = 0; i < NI; i++) ext += x_lds[b * NI + i] * wih_lds[c0 * NI + i];
      const float rec = rec_lds[b * COLS + c0];
      const float hn  = hst[k2] + DT_TAU * (-hst[k2] + rec + ext);
      const float htn = sigm(AG * (hn - ATH));
      const float rn  = rst[k2] + DT * ((1.0f - rst[k2]) * I_TAU_D
                                        - BETA * ust[k2] * rst[k2] * htst[k2]);
      const float un  = ust[k2] + DT * ((prl_[k2] - ust[k2]) * I_TAU_F
                                        + bprl[k2] * (1.0f - ust[k2]) * htst[k2]);
      const size_t ob = (size_t)b * T * H + (size_t)t * H + colg;
      o_ext[ob] = ext;
      o_h[ob]   = hn;
      o_r[ob]   = rn;
      o_u[ob]   = un;
      const size_t sx = (size_t)b * H + colg;
      sbW[sx] = (_Float16)(rn * un * psc_[k2] * htn);  // s_{t+1}
      htW[sx] = htn;
      hst[k2] = hn; htst[k2] = htn; rst[k2] = rn; ust[k2] = un;
      nst[k2] *= NDEC;
    }

    grid_barrier(bar);  // publish s_{t+1}, ht_{t+1}

    // ---- readout: this WG computes z[:, t, g] from full ht_{t+1} ----
    {
      const int b = tid >> 3, seg = tid & 7;
      const float4* htp = (const float4*)(htbuf + (size_t)(p ^ 1) * (B * H)
                                          + (size_t)b * H + seg * 128);
      const float4* wz  = (const float4*)(Whz + (size_t)g * H + seg * 128);
      float acc = 0.0f;
#pragma unroll 4
      for (int q = 0; q < 32; q++) {
        const float4 hv = htp[q];
        const float4 wv = wz[q];
        acc += hv.x * wv.x + hv.y * wv.y + hv.z * wv.z + hv.w * wv.w;
      }
      zred[tid] = acc;
      __syncthreads();
      if (tid < B) {
        float zz = offs[g];
#pragma unroll
        for (int q = 0; q < 8; q++) zz += zred[tid * 8 + q];
        o_z[(size_t)tid * T * NO + (size_t)t * NO + g] = zz;
      }
      __syncthreads();
    }
  }
}

extern "C" void kernel_launch(void* const* d_in, const int* in_sizes, int n_in,
                              void* d_out, int out_size, void* d_ws, size_t ws_size,
                              hipStream_t stream) {
  (void)in_sizes; (void)n_in; (void)out_size; (void)ws_size;
  const float* x   = (const float*)d_in[0];
  const float* h0  = (const float*)d_in[1];
  const float* r0  = (const float*)d_in[2];
  const float* u0  = (const float*)d_in[3];
  const float* Wih = (const float*)d_in[4];
  const float* Whh = (const float*)d_in[5];
  const float* Wmk = (const float*)d_in[6];
  const float* prl = (const float*)d_in[7];
  const float* psc = (const float*)d_in[8];
  const float* Whz = (const float*)d_in[9];
  const float* off = (const float*)d_in[10];

  float* out = (float*)d_out;
  const size_t BTH = (size_t)B * T * H;
  float* o_ext = out;
  float* o_h   = out + BTH;
  float* o_r   = out + 2 * BTH;
  float* o_u   = out + 3 * BTH;
  float* o_z   = out + 4 * BTH;

  char* ws = (char*)d_ws;
  _Float16* sbuf  = (_Float16*)ws;                                        // 64 KB
  float*    htbuf = (float*)(ws + 2 * (size_t)B * H * sizeof(_Float16));  // 128 KB
  unsigned* bar   = (unsigned*)(ws + 2 * (size_t)B * H * 2 + 2 * (size_t)B * H * 4);

  hipLaunchKernelGGL(init_bar, dim3(1), dim3(32), 0, stream, bar);
  hipLaunchKernelGGL(rnn_stp_persistent, dim3(NB), dim3(NT), 0, stream,
                     x, h0, r0, u0, Wih, Whh, Wmk, prl, psc, Whz, off,
                     o_ext, o_h, o_r, o_u, o_z, sbuf, htbuf, bar);
}
